// PointsConvolutionIntegrated_47571057770916
// MI455X (gfx1250) — matrix operand order repaired and checked by reading, not compiled, not run on hardware
//
#include <hip/hip_runtime.h>
#include <math.h>

typedef __attribute__((ext_vector_type(16))) _Float16 v16h;
typedef __attribute__((ext_vector_type(8)))  float    v8f;

#define WMMA_F16(a,b,c) __builtin_amdgcn_wmma_f32_16x16x32_f16(false,(a),false,(b),(short)0,(c),false,false)

constexpr int   NNODES  = 50000;
constexpr int   NEDGES  = 500000;
constexpr float PI_F    = 3.14159265358979f;
constexpr float INV_S8  = 0.35355339059327f;   // 1/sqrt(8)
constexpr float INV_S64 = 0.125f;              // 1/sqrt(64)
constexpr float INV_S32 = 0.17677669529664f;   // 1/sqrt(32)
constexpr float INV_S3  = 0.57735026918963f;   // 1/sqrt(3)

__device__ __forceinline__ float silu_f(float x) { return x / (1.0f + __expf(-x)); }

// ---- B-operand fragment load: packed [frag][lane][16] halves (32B/lane) ----
__device__ __forceinline__ v16h load_bfrag(const _Float16* __restrict__ p, int frag, int lane) {
    return *(const v16h*)(p + ((size_t)(frag * 32 + lane)) * 16);
}

// ---- A-operand fragment from an LDS row-major f16 matrix (16 x stride) ----
__device__ __forceinline__ v16h lds_afrag(const _Float16* h, int stride, int kt, int lane) {
    int m   = lane & 15;
    int off = (lane & 16) ? 8 : 0;
    v16h a;
#pragma unroll
    for (int j = 0; j < 8; ++j) a[j]     = h[m * stride + kt * 32 + off + j];
#pragma unroll
    for (int j = 0; j < 8; ++j) a[8 + j] = h[m * stride + kt * 32 + 16 + off + j];
    return a;
}

// =====================================================================
// Pack a KxN row-major f32 weight matrix into f16 WMMA B fragments.
// grid.x = KT*NT fragments, 32 threads/block.
// =====================================================================
__global__ void __launch_bounds__(32) pack_b(const float* __restrict__ W,
                                             _Float16* __restrict__ dst,
                                             int Kreal, int N, int NT) {
    int frag = blockIdx.x;
    int nt   = frag % NT;
    int kt   = frag / NT;
    int lane = threadIdx.x;
    int n     = nt * 16 + (lane & 15);
    int kbase = kt * 32 + ((lane & 16) ? 8 : 0);
#pragma unroll
    for (int j = 0; j < 16; ++j) {
        int k = kbase + (j < 8 ? j : 8 + j);
        float v = (k < Kreal) ? W[k * N + n] : 0.0f;
        dst[((size_t)(frag * 32 + lane)) * 16 + j] = (_Float16)v;
    }
}

__global__ void zero_f32(float* __restrict__ p, int n) {
    int i = blockIdx.x * blockDim.x + threadIdx.x;
    int stride = gridDim.x * blockDim.x;
    for (; i < n; i += stride) p[i] = 0.0f;
}

// =====================================================================
// Node projections: sc_s/sc_v and l1_s/l1_v  (fctp with inv = 1/sqrt(32))
// One wave handles 16 nodes. 16 WMMAs per tile.
// =====================================================================
__global__ void __launch_bounds__(32) node_proj(
    const float* __restrict__ xin, const float* __restrict__ natt,
    const _Float16* __restrict__ scws, const _Float16* __restrict__ scwv,
    const _Float16* __restrict__ l1ws, const _Float16* __restrict__ l1wv,
    float* __restrict__ sc_s, float* __restrict__ sc_v,
    float* __restrict__ l1_s, float* __restrict__ l1_v) {
    __shared__ float sa[16];
    int lane = threadIdx.x;
    int tile = blockIdx.x;
    int m    = lane & 15;
    int offA = (lane & 16) ? 8 : 0;
    int col  = m;
    int node = tile * 16 + m;

    if (lane < 16) sa[lane] = natt[tile * 16 + lane];
    __syncthreads();

    // ---- scalar channel A fragment: x_s = xin[:, 0:32], K = 32 ----
    v16h as;
#pragma unroll
    for (int j = 0; j < 8; ++j) as[j]     = (_Float16)xin[node * 128 + offA + j];
#pragma unroll
    for (int j = 0; j < 8; ++j) as[8 + j] = (_Float16)xin[node * 128 + 16 + offA + j];

#pragma unroll
    for (int nt = 0; nt < 2; ++nt) {
        v8f c0 = {};
        c0 = WMMA_F16(as, load_bfrag(scws, nt, lane), c0);
        v8f c1 = {};
        c1 = WMMA_F16(as, load_bfrag(l1ws, nt, lane), c1);
#pragma unroll
        for (int r = 0; r < 8; ++r) {
            int mo = r + offA;
            int no = tile * 16 + mo;
            float sc = INV_S32 * sa[mo];
            sc_s[no * 32 + nt * 16 + col] = c0[r] * sc;
            l1_s[no * 32 + nt * 16 + col] = c1[r] * sc;
        }
    }

    // ---- vector channels: x_v[n][k][mc] = xin[n][32 + 3k + mc] ----
#pragma unroll
    for (int mc = 0; mc < 3; ++mc) {
        v16h av;
#pragma unroll
        for (int j = 0; j < 8; ++j) av[j]     = (_Float16)xin[node * 128 + 32 + (offA + j) * 3 + mc];
#pragma unroll
        for (int j = 0; j < 8; ++j) av[8 + j] = (_Float16)xin[node * 128 + 32 + (16 + offA + j) * 3 + mc];
#pragma unroll
        for (int nt = 0; nt < 2; ++nt) {
            v8f c0 = {};
            c0 = WMMA_F16(av, load_bfrag(scwv, nt, lane), c0);
            v8f c1 = {};
            c1 = WMMA_F16(av, load_bfrag(l1wv, nt, lane), c1);
#pragma unroll
            for (int r = 0; r < 8; ++r) {
                int mo = r + offA;
                int no = tile * 16 + mo;
                float sc = INV_S32 * sa[mo];
                int cc = nt * 16 + col;
                sc_v[no * 96 + cc * 3 + mc] = c0[r] * sc;
                l1_v[no * 96 + cc * 3 + mc] = c1[r] * sc;
            }
        }
    }
}

// =====================================================================
// Edge kernel: RBF -> 3-layer MLP (28 WMMAs) -> message scatter (atomics)
// One wave handles 16 edges.
// =====================================================================
__global__ void __launch_bounds__(32) edge_kernel(
    const float* __restrict__ dist, const float* __restrict__ sh_attr,
    const int* __restrict__ esrc, const int* __restrict__ edst,
    const _Float16* __restrict__ w1p, const _Float16* __restrict__ w2p,
    const _Float16* __restrict__ w3p,
    const float* __restrict__ l1s, const float* __restrict__ l1v,
    float* __restrict__ aggs, float* __restrict__ aggv, float* __restrict__ cnt) {
    __shared__ _Float16 h1[16 * 64];
    __shared__ _Float16 h2[16 * 64];
    __shared__ float ssh[16 * 4];
    __shared__ int   ssrc[16];
    __shared__ int   sdst[16];

    int lane = threadIdx.x;
    int tile = blockIdx.x;
    int col  = lane & 15;
    int offA = (lane & 16) ? 8 : 0;

    // ---- RBF features as A fragment (K padded 8 -> 32 with zeros) ----
    v16h aR = {};
    if (lane < 16) {
        int e = tile * 16 + lane;
        float d = dist[e];
        ssrc[lane] = esrc[e];
        int dd = edst[e];
        sdst[lane] = dd;
        ssh[lane * 4 + 0] = sh_attr[e * 4 + 0];
        ssh[lane * 4 + 1] = sh_attr[e * 4 + 1];
        ssh[lane * 4 + 2] = sh_attr[e * 4 + 2];
        ssh[lane * 4 + 3] = sh_attr[e * 4 + 3];
        float cut = (d <= 6.0f) ? 0.5f * (__cosf(PI_F * d / 6.0f) + 1.0f) : 0.0f;
#pragma unroll
        for (int i = 0; i < 8; ++i) {
            float fd = d * (float)(i + 1) * (PI_F / 6.0f);
            float s  = (fabsf(fd) < 1e-6f) ? 1.0f : (__sinf(fd) / fd);
            aR[i] = (_Float16)(cut * s);
        }
        atomicAdd(&cnt[dd], 1.0f);
    }
    __syncthreads();

    // ---- layer 1: (16x32)x(32x64), silu(c/sqrt(8)) -> h1 ----
#pragma unroll
    for (int nt = 0; nt < 4; ++nt) {
        v8f c = {};
        c = WMMA_F16(aR, load_bfrag(w1p, nt, lane), c);
#pragma unroll
        for (int r = 0; r < 8; ++r) {
            int mo = r + offA;
            h1[mo * 64 + nt * 16 + col] = (_Float16)silu_f(c[r] * INV_S8);
        }
    }
    __syncthreads();

    // ---- layer 2: (16x64)x(64x64), silu(c/8) -> h2 ----
    {
        v16h a0 = lds_afrag(h1, 64, 0, lane);
        v16h a1 = lds_afrag(h1, 64, 1, lane);
#pragma unroll
        for (int nt = 0; nt < 4; ++nt) {
            v8f c = {};
            c = WMMA_F16(a0, load_bfrag(w2p, 0 * 4 + nt, lane), c);
            c = WMMA_F16(a1, load_bfrag(w2p, 1 * 4 + nt, lane), c);
#pragma unroll
            for (int r = 0; r < 8; ++r) {
                int mo = r + offA;
                h2[mo * 64 + nt * 16 + col] = (_Float16)silu_f(c[r] * INV_S64);
            }
        }
    }
    __syncthreads();

    // ---- layer 3: (16x64)x(64x128) -> weight, fused message scatter ----
    v16h a0 = lds_afrag(h2, 64, 0, lane);
    v16h a1 = lds_afrag(h2, 64, 1, lane);
#pragma unroll
    for (int nt = 0; nt < 8; ++nt) {
        v8f c = {};
        c = WMMA_F16(a0, load_bfrag(w3p, 0 * 8 + nt, lane), c);
        c = WMMA_F16(a1, load_bfrag(w3p, 1 * 8 + nt, lane), c);
        int cc  = (nt & 1) * 16 + col;   // channel within a 32-wide weight block
        int blk = nt >> 1;               // 0:w00 1:w01 2:w10 3:w11
#pragma unroll
        for (int r = 0; r < 8; ++r) {
            int m   = r + offA;
            float w = c[r] * INV_S64;
            int src = ssrc[m];
            int dst = sdst[m];
            float s0 = ssh[m * 4 + 0];
            if (blk == 0) {             // o00 -> agg_s[:, cc]
                float g = l1s[src * 32 + cc];
                atomicAdd(&aggs[dst * 64 + cc], w * g * s0);
            } else if (blk == 1) {      // o01 -> agg_v[:, cc, :]
                float wg = w * l1s[src * 32 + cc];
                int base = dst * 192 + cc * 3;
                atomicAdd(&aggv[base + 0], wg * ssh[m * 4 + 1]);
                atomicAdd(&aggv[base + 1], wg * ssh[m * 4 + 2]);
                atomicAdd(&aggv[base + 2], wg * ssh[m * 4 + 3]);
            } else if (blk == 2) {      // o10 -> agg_v[:, 32+cc, :]
                int gbase = src * 96 + cc * 3;
                int base  = dst * 192 + (32 + cc) * 3;
                atomicAdd(&aggv[base + 0], w * l1v[gbase + 0] * s0);
                atomicAdd(&aggv[base + 1], w * l1v[gbase + 1] * s0);
                atomicAdd(&aggv[base + 2], w * l1v[gbase + 2] * s0);
            } else {                    // o11 -> agg_s[:, 32+cc]
                int gbase = src * 96 + cc * 3;
                float dot = l1v[gbase + 0] * ssh[m * 4 + 1] +
                            l1v[gbase + 1] * ssh[m * 4 + 2] +
                            l1v[gbase + 2] * ssh[m * 4 + 3];
                atomicAdd(&aggs[dst * 64 + 32 + cc], w * dot * INV_S3);
            }
        }
    }
}

// =====================================================================
// Output stage: lin2 (16 WMMAs/tile), alpha gate, final combine.
// =====================================================================
__global__ void __launch_bounds__(32) node_out(
    const float* __restrict__ natt,
    const _Float16* __restrict__ l2ws, const _Float16* __restrict__ l2wv,
    const float* __restrict__ alpha_w,
    const float* __restrict__ sc_s, const float* __restrict__ sc_v,
    const float* __restrict__ aggs, const float* __restrict__ aggv,
    const float* __restrict__ cnt, float* __restrict__ out) {
    __shared__ float salpha[16];
    __shared__ float snatt[16];
    __shared__ float scnt[16];

    int lane = threadIdx.x;
    int tile = blockIdx.x;
    int m    = lane & 15;
    int offA = (lane & 16) ? 8 : 0;
    int col  = m;
    int node = tile * 16 + m;

    if (lane < 16) {
        int n = tile * 16 + lane;
        float cv = cnt[n];
        cv = (cv < 1.0f) ? 1.0f : cv;
        float rc = 1.0f / cv;
        float av = natt[n];
        scnt[lane]  = rc;
        snatt[lane] = av;
        float dot = 0.0f;
        for (int k = 0; k < 64; ++k) dot += aggs[n * 64 + k] * alpha_w[k];
        salpha[lane] = tanhf(dot * rc * av * INV_S64);
    }
    __syncthreads();

    float rc = scnt[m];

    // ---- l2_s: (agg_s/cnt) @ lin2_ws ----
    v16h a0, a1;
#pragma unroll
    for (int j = 0; j < 8; ++j) a0[j]     = (_Float16)(aggs[node * 64 + offA + j] * rc);
#pragma unroll
    for (int j = 0; j < 8; ++j) a0[8 + j] = (_Float16)(aggs[node * 64 + 16 + offA + j] * rc);
#pragma unroll
    for (int j = 0; j < 8; ++j) a1[j]     = (_Float16)(aggs[node * 64 + 32 + offA + j] * rc);
#pragma unroll
    for (int j = 0; j < 8; ++j) a1[8 + j] = (_Float16)(aggs[node * 64 + 48 + offA + j] * rc);

#pragma unroll
    for (int nt = 0; nt < 2; ++nt) {
        v8f c = {};
        c = WMMA_F16(a0, load_bfrag(l2ws, 0 * 2 + nt, lane), c);
        c = WMMA_F16(a1, load_bfrag(l2ws, 1 * 2 + nt, lane), c);
#pragma unroll
        for (int r = 0; r < 8; ++r) {
            int mo = r + offA;
            int no = tile * 16 + mo;
            int cc = nt * 16 + col;
            float l2 = c[r] * INV_S64 * snatt[mo];
            out[no * 128 + cc] = sc_s[no * 32 + cc] + salpha[mo] * l2;
        }
    }

    // ---- l2_v: per-component (agg_v/cnt) @ lin2_wv ----
#pragma unroll
    for (int mc = 0; mc < 3; ++mc) {
        v16h b0, b1;
#pragma unroll
        for (int j = 0; j < 8; ++j) b0[j]     = (_Float16)(aggv[node * 192 + (offA + j) * 3 + mc] * rc);
#pragma unroll
        for (int j = 0; j < 8; ++j) b0[8 + j] = (_Float16)(aggv[node * 192 + (16 + offA + j) * 3 + mc] * rc);
#pragma unroll
        for (int j = 0; j < 8; ++j) b1[j]     = (_Float16)(aggv[node * 192 + (32 + offA + j) * 3 + mc] * rc);
#pragma unroll
        for (int j = 0; j < 8; ++j) b1[8 + j] = (_Float16)(aggv[node * 192 + (48 + offA + j) * 3 + mc] * rc);

#pragma unroll
        for (int nt = 0; nt < 2; ++nt) {
            v8f c = {};
            c = WMMA_F16(b0, load_bfrag(l2wv, 0 * 2 + nt, lane), c);
            c = WMMA_F16(b1, load_bfrag(l2wv, 1 * 2 + nt, lane), c);
#pragma unroll
            for (int r = 0; r < 8; ++r) {
                int mo = r + offA;
                int no = tile * 16 + mo;
                int cc = nt * 16 + col;
                float l2 = c[r] * INV_S64 * snatt[mo];
                out[no * 128 + 32 + cc * 3 + mc] = sc_v[no * 96 + cc * 3 + mc] + salpha[mo] * l2;
            }
        }
    }
}

// =====================================================================
extern "C" void kernel_launch(void* const* d_in, const int* in_sizes, int n_in,
                              void* d_out, int out_size, void* d_ws, size_t ws_size,
                              hipStream_t stream) {
    const float* node_input = (const float*)d_in[0];
    const float* node_attr  = (const float*)d_in[1];
    const float* edge_sh    = (const float*)d_in[2];
    const float* edge_dist  = (const float*)d_in[3];
    const int*   edge_src   = (const int*)d_in[4];
    const int*   edge_dst   = (const int*)d_in[5];
    const float* sc_ws   = (const float*)d_in[6];
    const float* sc_wv   = (const float*)d_in[7];
    const float* lin1_ws = (const float*)d_in[8];
    const float* lin1_wv = (const float*)d_in[9];
    const float* fc_w1   = (const float*)d_in[10];
    const float* fc_w2   = (const float*)d_in[11];
    const float* fc_w3   = (const float*)d_in[12];
    const float* lin2_ws = (const float*)d_in[13];
    const float* lin2_wv = (const float*)d_in[14];
    const float* alpha_w = (const float*)d_in[15];
    float* out = (float*)d_out;

    char* ws = (char*)d_ws;
    _Float16* w1p   = (_Float16*)(ws + 0);        //  4 frags * 1024B
    _Float16* w2p   = (_Float16*)(ws + 4096);     //  8 frags
    _Float16* w3p   = (_Float16*)(ws + 12288);    // 16 frags
    _Float16* scwsp = (_Float16*)(ws + 28672);    //  2 frags
    _Float16* scwvp = (_Float16*)(ws + 30720);
    _Float16* l1wsp = (_Float16*)(ws + 32768);
    _Float16* l1wvp = (_Float16*)(ws + 34816);
    _Float16* l2wsp = (_Float16*)(ws + 36864);    //  4 frags
    _Float16* l2wvp = (_Float16*)(ws + 40960);

    float* fbase = (float*)(ws + 45056);
    float* l1_s  = fbase;
    float* l1_v  = l1_s + (size_t)NNODES * 32;
    float* sc_s  = l1_v + (size_t)NNODES * 96;
    float* sc_v  = sc_s + (size_t)NNODES * 32;
    float* agg_s = sc_v + (size_t)NNODES * 96;
    float* agg_v = agg_s + (size_t)NNODES * 64;
    float* cnt   = agg_v + (size_t)NNODES * 192;

    // ---- pack weights into f16 B fragments ----
    pack_b<<<4, 32, 0, stream>>>(fc_w1, w1p, 8, 64, 4);
    pack_b<<<8, 32, 0, stream>>>(fc_w2, w2p, 64, 64, 4);
    pack_b<<<16, 32, 0, stream>>>(fc_w3, w3p, 64, 128, 8);
    pack_b<<<2, 32, 0, stream>>>(sc_ws, scwsp, 32, 32, 2);
    pack_b<<<2, 32, 0, stream>>>(sc_wv, scwvp, 32, 32, 2);
    pack_b<<<2, 32, 0, stream>>>(lin1_ws, l1wsp, 32, 32, 2);
    pack_b<<<2, 32, 0, stream>>>(lin1_wv, l1wvp, 32, 32, 2);
    pack_b<<<4, 32, 0, stream>>>(lin2_ws, l2wsp, 64, 32, 2);
    pack_b<<<4, 32, 0, stream>>>(lin2_wv, l2wvp, 64, 32, 2);

    // ---- zero agg_s | agg_v | cnt (contiguous) ----
    zero_f32<<<1024, 256, 0, stream>>>(agg_s, NNODES * (64 + 192 + 1));

    // ---- node projections ----
    node_proj<<<NNODES / 16, 32, 0, stream>>>(node_input, node_attr,
                                              scwsp, scwvp, l1wsp, l1wvp,
                                              sc_s, sc_v, l1_s, l1_v);

    // ---- edge MLP + message scatter ----
    edge_kernel<<<NEDGES / 16, 32, 0, stream>>>(edge_dist, edge_sh, edge_src, edge_dst,
                                                w1p, w2p, w3p, l1_s, l1_v,
                                                agg_s, agg_v, cnt);

    // ---- output stage ----
    node_out<<<NNODES / 16, 32, 0, stream>>>(node_attr, l2wsp, l2wvp, alpha_w,
                                             sc_s, sc_v, agg_s, agg_v, cnt, out);
}
